// ConvolutionCalculator_70368744177863
// MI455X (gfx1250) — matrix operand order repaired and checked by reading, not compiled, not run on hardware
//
#include <hip/hip_runtime.h>

typedef __attribute__((ext_vector_type(16))) _Float16 v16h;
typedef __attribute__((ext_vector_type(8)))  float    v8f;
typedef __attribute__((ext_vector_type(4)))  float    v4f;

#define NNODES  10000
#define NEDGES  640000
#define NCH     64
#define NTILES  (NEDGES / 16)
#define C_SH       0.4886025119029199f
#define INV_SQRT2  0.7071067811865476f

// D[M=channel, N=edge] = A(W2^T)[16x32] x B(h^T)[32x16] per tile of 16 edges.
// Each lane owns one edge (N = lane&15) and 8 consecutive channels per m-tile.
// Streaming x_j/out use non-temporal hints so the 192MB L2 stays reserved for
// the 400MB random-gather distances table.
__global__ __launch_bounds__(256)
void e3conv_wmma_kernel(const float* __restrict__ x_j,        // [E,3,C]
                        const float* __restrict__ positions,  // [N,3]
                        const float* __restrict__ distances,  // [N,N]
                        const float* __restrict__ W1,         // [1,H]
                        const float* __restrict__ b1,         // [H]
                        const float* __restrict__ W2,         // [H,C]
                        const float* __restrict__ b2,         // [C]
                        const long long* __restrict__ eidx,   // [2,E] int64
                        float* __restrict__ out)              // [E,3,C]
{
    const int lane = threadIdx.x & 31;
    const int hi   = (lane >> 4) & 1;   // 0: lanes 0-15, 1: lanes 16-31
    const int lrow = lane & 15;
    const int wavesPerBlock = blockDim.x >> 5;
    const int gwave = blockIdx.x * wavesPerBlock + (threadIdx.x >> 5);
    const int nwave = gridDim.x * wavesPerBlock;

    // ---- per-lane radial-MLP weights for the B (h^T) fragment ----
    // B fragment (32x16 f16): element t -> K = ((t<8)? t : t+8) + hi*8 (+32 for 2nd K-piece)
    float w1v[32], b1v[32];
#pragma unroll
    for (int p = 0; p < 2; ++p) {
#pragma unroll
        for (int t = 0; t < 16; ++t) {
            const int K = ((t < 8) ? t : t + 8) + hi * 8 + p * 32;
            w1v[p * 16 + t] = W1[K];
            b1v[p * 16 + t] = b1[K];
        }
    }

    // ---- A fragments = W2^T tiles (M=channel, K=hidden), resident in VGPRs ----
    // A fragment (16x32 f16): element t -> K = ((t<8)? t : t+8) + hi*8 (+32)
    v16h afrag[8];
#pragma unroll
    for (int m = 0; m < 4; ++m) {
#pragma unroll
        for (int p = 0; p < 2; ++p) {
            v16h a;
#pragma unroll
            for (int t = 0; t < 16; ++t) {
                const int K = ((t < 8) ? t : t + 8) + hi * 8 + p * 32;
                a[t] = (_Float16)W2[K * NCH + m * 16 + lrow]; // A[c,k] = W2[k,c]
            }
            afrag[m * 2 + p] = a;
        }
    }

    // ---- per-lane bias (loop-invariant): 8 channels per m-tile ----
    v4f bb[8];
#pragma unroll
    for (int m = 0; m < 4; ++m) {
        const int cb = m * 16 + hi * 8;
        bb[2 * m + 0] = *(const v4f*)(b2 + cb);
        bb[2 * m + 1] = *(const v4f*)(b2 + cb + 4);
    }

    for (int tile = gwave; tile < NTILES; tile += nwave) {
        const int e = tile * 16 + lrow;          // lanes l and l+16: same edge
        const long long ii = eidx[e];
        const long long jj = eidx[NEDGES + e];
        const float d    = distances[ii * (long long)NNODES + jj];
        const float invd = 1.0f / d;
        const float vx = (positions[ii * 3 + 0] - positions[jj * 3 + 0]) * invd;
        const float vy = (positions[ii * 3 + 1] - positions[jj * 3 + 1]) * invd;
        const float vz = (positions[ii * 3 + 2] - positions[jj * 3 + 2]) * invd;
        const float sh0 = C_SH * vy;             // m = -1
        const float sh1 = C_SH * vz;             // m =  0
        const float sh2 = C_SH * vx;             // m = +1

        // ---- B fragments: h[e,k] = relu(d*W1[k]+b1[k]), f16 ----
        v16h bf0, bf1;
#pragma unroll
        for (int t = 0; t < 16; ++t) {
            float h0 = fmaf(d, w1v[t],      b1v[t]);
            float h1 = fmaf(d, w1v[16 + t], b1v[16 + t]);
            h0 = h0 > 0.0f ? h0 : 0.0f;
            h1 = h1 > 0.0f ? h1 : 0.0f;
            bf0[t] = (_Float16)h0;
            bf1[t] = (_Float16)h1;
        }

        const float* xb = x_j + (size_t)e * 192;
        float*       ob = out + (size_t)e * 192;

#pragma unroll
        for (int m = 0; m < 4; ++m) {
            v8f acc = {};
            acc = __builtin_amdgcn_wmma_f32_16x16x32_f16(
                      false, afrag[2 * m + 0], false, bf0, (short)0, acc, false, false);
            acc = __builtin_amdgcn_wmma_f32_16x16x32_f16(
                      false, afrag[2 * m + 1], false, bf1, (short)0, acc, false, false);

            const int cb = m * 16 + hi * 8;      // 8 consecutive channels per lane
            const v4f x0a = __builtin_nontemporal_load((const v4f*)(xb + cb));
            const v4f x0b = __builtin_nontemporal_load((const v4f*)(xb + cb + 4));
            const v4f x1a = __builtin_nontemporal_load((const v4f*)(xb + 64 + cb));
            const v4f x1b = __builtin_nontemporal_load((const v4f*)(xb + 64 + cb + 4));
            const v4f x2a = __builtin_nontemporal_load((const v4f*)(xb + 128 + cb));
            const v4f x2b = __builtin_nontemporal_load((const v4f*)(xb + 128 + cb + 4));

            v4f o0a, o0b, o1a, o1b, o2a, o2b;
#pragma unroll
            for (int s = 0; s < 4; ++s) {
                const float sc0 = INV_SQRT2 * (acc[s]     + bb[2 * m + 0][s]);
                const float sc1 = INV_SQRT2 * (acc[4 + s] + bb[2 * m + 1][s]);
                o0a[s] = sc0 * (sh1 * x2a[s] - sh2 * x1a[s]);
                o1a[s] = sc0 * (sh2 * x0a[s] - sh0 * x2a[s]);
                o2a[s] = sc0 * (sh0 * x1a[s] - sh1 * x0a[s]);
                o0b[s] = sc1 * (sh1 * x2b[s] - sh2 * x1b[s]);
                o1b[s] = sc1 * (sh2 * x0b[s] - sh0 * x2b[s]);
                o2b[s] = sc1 * (sh0 * x1b[s] - sh1 * x0b[s]);
            }
            __builtin_nontemporal_store(o0a, (v4f*)(ob + cb));
            __builtin_nontemporal_store(o0b, (v4f*)(ob + cb + 4));
            __builtin_nontemporal_store(o1a, (v4f*)(ob + 64 + cb));
            __builtin_nontemporal_store(o1b, (v4f*)(ob + 64 + cb + 4));
            __builtin_nontemporal_store(o2a, (v4f*)(ob + 128 + cb));
            __builtin_nontemporal_store(o2b, (v4f*)(ob + 128 + cb + 4));
        }
    }
}

extern "C" void kernel_launch(void* const* d_in, const int* in_sizes, int n_in,
                              void* d_out, int out_size, void* d_ws, size_t ws_size,
                              hipStream_t stream) {
    const float*     x_j       = (const float*)d_in[0];
    const float*     positions = (const float*)d_in[1];
    const float*     distances = (const float*)d_in[2];
    const float*     W1        = (const float*)d_in[3];
    const float*     b1        = (const float*)d_in[4];
    const float*     W2        = (const float*)d_in[5];
    const float*     b2        = (const float*)d_in[6];
    const long long* eidx      = (const long long*)d_in[7];
    float*           out       = (float*)d_out;

    // 1250 blocks x 8 waves = 10000 waves; 40000 tiles -> 4 tiles/wave,
    // amortizing the resident W2^T WMMA fragments across the grid-stride loop.
    dim3 grid(1250), block(256);
    e3conv_wmma_kernel<<<grid, block, 0, stream>>>(
        x_j, positions, distances, W1, b1, W2, b2, eidx, out);
}
